// cubic_attention_72773925864316
// MI455X (gfx1250) — compile-verified, matrix operand order checked by reading, not run on hardware
//
#include <hip/hip_runtime.h>
#include <hip/hip_bf16.h>

// Problem constants (match the reference)
#define NB   8
#define DIM  64
#define HH   256
#define WW   256
#define KK   3
#define EPSV 1e-5f

typedef __attribute__((ext_vector_type(16))) _Float16 v16h;
typedef __attribute__((ext_vector_type(8)))  float    v8f;
typedef __attribute__((ext_vector_type(4)))  float    f4;   // native vec for NT stores

// ---------------------------------------------------------------------------
// Kernel 1: global average pool.  gap[n*DIM+c] = mean(x[n,c,:,:])
// One block per (n,c); 256 threads; 128-bit coalesced loads; LDS reduction.
// Also pulls x into L2 (x = 128MB < 192MB L2) for the conv pass.
// ---------------------------------------------------------------------------
__global__ void gap_kernel(const float* __restrict__ x, float* __restrict__ gap) {
    __shared__ float red[256];
    const int nc = blockIdx.x;                     // 0..511
    const f4* xr = (const f4*)(x + (size_t)nc * (HH * WW));
    float s = 0.f;
    for (int i = threadIdx.x; i < (HH * WW) / 4; i += 256) {
        f4 v = xr[i];
        s += v.x + v.y + v.z + v.w;
    }
    red[threadIdx.x] = s;
    __syncthreads();
    for (int off = 128; off > 0; off >>= 1) {
        if (threadIdx.x < off) red[threadIdx.x] += red[threadIdx.x + off];
        __syncthreads();
    }
    if (threadIdx.x == 0) gap[nc] = red[0] * (1.0f / (HH * WW));
}

// ---------------------------------------------------------------------------
// Kernel 2: taps = tanh(BN(gap @ w^T)) for both branches, via WMMA.
// GEMM: A = gap (M=16 padded x K=64), B = w^T (K=64 x N=192), per branch.
// 24 waves (wave32) in one block: wave -> (branch, 16-col tile of N=192).
// Each wave: two v_wmma_f32_16x16x32_f16 along K, then fused BN + tanh.
//
// 16-bit A 16x32 layout (ISA 7.12.2): lane L holds row M=L%16; element e of
// the v16h maps to K = 16*(e/8) + 8*(L/16) + (e%8).  B (32x16) is the
// transpose-symmetric layout: lane L holds column N=L%16, same K mapping.
// C/D f32: VGPR r, lane L -> (M = r + 8*(L/16), N = L%16).
// ---------------------------------------------------------------------------
__global__ void taps_kernel(const float* __restrict__ gap,
                            const float* __restrict__ wHp, const float* __restrict__ gH,
                            const float* __restrict__ bH,  const float* __restrict__ mH,
                            const float* __restrict__ vH,
                            const float* __restrict__ wWp, const float* __restrict__ gW,
                            const float* __restrict__ bW,  const float* __restrict__ mW,
                            const float* __restrict__ vW,
                            float* __restrict__ tapsH, float* __restrict__ tapsW) {
    const int wave   = threadIdx.x >> 5;   // 0..23
    const int lane   = threadIdx.x & 31;
    const int branch = wave / 12;          // 0 = H branch, 1 = W branch
    const int tile   = wave % 12;          // 16-wide tile of N=192
    const int M      = lane & 15;          // A row / C col-group selector
    const int hg     = lane >> 4;          // lane half-group
    const int j      = tile * 16 + (lane & 15);   // output column 0..191

    const float* wmat = branch ? wWp : wHp;

    v16h a0, a1, b0, b1;
#pragma unroll
    for (int e = 0; e < 16; ++e) {
        const int k = ((e >> 3) << 4) + (hg << 3) + (e & 7);   // K in [0,32)
        const float av0 = (M < NB) ? gap[M * DIM + k]      : 0.f;
        const float av1 = (M < NB) ? gap[M * DIM + 32 + k] : 0.f;
        a0[e] = (_Float16)av0;
        a1[e] = (_Float16)av1;
        b0[e] = (_Float16)wmat[j * DIM + k];        // B[k][j] = w[j][k]
        b1[e] = (_Float16)wmat[j * DIM + 32 + k];
    }

    v8f acc = {};
    // (neg_a, A, neg_b, B, c_mod, C, reuse_a, reuse_b)
    acc = __builtin_amdgcn_wmma_f32_16x16x32_f16(false, a0, false, b0,
                                                 (short)0, acc, false, false);
    acc = __builtin_amdgcn_wmma_f32_16x16x32_f16(false, a1, false, b1,
                                                 (short)0, acc, false, false);

    const float gj = branch ? gW[j] : gH[j];
    const float bj = branch ? bW[j] : bH[j];
    const float mj = branch ? mW[j] : mH[j];
    const float vj = branch ? vW[j] : vH[j];
    const float scale = gj * rsqrtf(vj + EPSV);
    float* taps = branch ? tapsW : tapsH;

#pragma unroll
    for (int r = 0; r < 8; ++r) {
        const int row = r + (hg << 3);             // M index 0..15
        if (row < NB) {
            const float f = (acc[r] - mj) * scale + bj;
            taps[row * (DIM * KK) + j] = tanhf(f); // j = c*3 + k
        }
    }
}

// ---------------------------------------------------------------------------
// Kernel 3: fused depthwise 3-tap strip conv, both branches.
// 2 * 8*64*256*64 vec4 tasks; branch bit (bit 23) is uniform per block.
// x reads use default RT policy (hit L2; x is resident). Output stores are
// NON-TEMPORAL (global_store_b128 th:TH_STORE_NT) so the 256MB output stream
// doesn't evict x from L2 -> HBM traffic ~128MB read + 256MB write.
// out layout: (n, 2*DIM, H, W), branch 0 -> ch [0,64), branch 1 -> [64,128).
// ---------------------------------------------------------------------------
__global__ void strip_conv_kernel(const float* __restrict__ x,
                                  const float* __restrict__ tapsH,
                                  const float* __restrict__ tapsW,
                                  float* __restrict__ out) {
    const unsigned t = blockIdx.x * blockDim.x + threadIdx.x;   // < 2^24
    const int w4     = t & 63;
    const int h      = (t >> 6) & 255;
    const int c      = (t >> 14) & 63;
    const int n      = (t >> 20) & 7;
    const int branch = (t >> 23) & 1;

    const float* tp = (branch ? tapsW : tapsH) + (n * DIM + c) * KK;
    const float t0 = tp[0], t1 = tp[1], t2 = tp[2];

    const float* xb = x + ((size_t)(n * DIM + c)) * (HH * WW);
    const int w0 = w4 << 2;
    f4 o;

    if (branch == 0) {
        // taps along W, reflect pad on width
        const f4 v = *(const f4*)(xb + h * WW + w0);
        const float left  = (w0 == 0)      ? v.y : xb[h * WW + w0 - 1];  // x[1]   when w0==0
        const float right = (w0 == WW - 4) ? v.z : xb[h * WW + w0 + 4];  // x[254] when w0==252
        o.x = t0 * left + t1 * v.x + t2 * v.y;
        o.y = t0 * v.x  + t1 * v.y + t2 * v.z;
        o.z = t0 * v.y  + t1 * v.z + t2 * v.w;
        o.w = t0 * v.z  + t1 * v.w + t2 * right;
    } else {
        // taps along H, reflect pad on height
        const int hm = (h == 0)      ? 1      : h - 1;
        const int hp = (h == HH - 1) ? HH - 2 : h + 1;
        const f4 a = *(const f4*)(xb + hm * WW + w0);
        const f4 b = *(const f4*)(xb + h  * WW + w0);
        const f4 d = *(const f4*)(xb + hp * WW + w0);
        o.x = t0 * a.x + t1 * b.x + t2 * d.x;
        o.y = t0 * a.y + t1 * b.y + t2 * d.y;
        o.z = t0 * a.z + t1 * b.z + t2 * d.z;
        o.w = t0 * a.w + t1 * b.w + t2 * d.w;
    }

    f4* ob = (f4*)(out + (((size_t)(n * 2 * DIM + branch * DIM + c)) * HH + h) * WW + w0);
    __builtin_nontemporal_store(o, ob);   // keep x resident in L2
}

// ---------------------------------------------------------------------------
// Launch
// ---------------------------------------------------------------------------
extern "C" void kernel_launch(void* const* d_in, const int* in_sizes, int n_in,
                              void* d_out, int out_size, void* d_ws, size_t ws_size,
                              hipStream_t stream) {
    const float* x  = (const float*)d_in[0];
    const float* wH = (const float*)d_in[1];
    const float* gH = (const float*)d_in[2];
    const float* bH = (const float*)d_in[3];
    const float* mH = (const float*)d_in[4];
    const float* vH = (const float*)d_in[5];
    const float* wW = (const float*)d_in[6];
    const float* gW = (const float*)d_in[7];
    const float* bW = (const float*)d_in[8];
    const float* mW = (const float*)d_in[9];
    const float* vW = (const float*)d_in[10];
    float* out = (float*)d_out;

    // Workspace layout (floats): gap[512] | tapsH[8*192] | tapsW[8*192]
    float* ws    = (float*)d_ws;
    float* gap   = ws;
    float* tapsH = ws + NB * DIM;                    // 512
    float* tapsW = tapsH + NB * DIM * KK;            // +1536

    // Phase 1: GAP (reads x once; warms L2 for phase 3)
    gap_kernel<<<NB * DIM, 256, 0, stream>>>(x, gap);

    // Phase 2: tiny GEMM (WMMA) + BN + tanh -> dynamic taps, both branches
    taps_kernel<<<1, 24 * 32, 0, stream>>>(gap,
                                           wH, gH, bH, mH, vH,
                                           wW, gW, bW, mW, vW,
                                           tapsH, tapsW);

    // Phase 3: fused depthwise strip conv, both branches, NT output stores
    // tasks = 2 * N * DIM * H * (W/4) = 2^24 threads
    const unsigned total  = 2u * NB * DIM * HH * (WW / 4);
    const unsigned blocks = total / 256;
    strip_conv_kernel<<<blocks, 256, 0, stream>>>(x, tapsH, tapsW, out);
}
